// WanSelfAttention_29472065585336
// MI455X (gfx1250) — matrix operand order repaired and checked by reading, not compiled, not run on hardware
//
#include <hip/hip_runtime.h>

typedef __bf16 bf16;
typedef __attribute__((ext_vector_type(16))) __bf16 v16bf;
typedef __attribute__((ext_vector_type(8)))  float  v8f;
typedef unsigned int uint;
typedef __attribute__((ext_vector_type(4))) uint u32x4;
typedef __attribute__((ext_vector_type(4))) int  i32x4;
typedef __attribute__((ext_vector_type(8))) int  i32x8;

#define DIM     1536
#define NHEADS  12
#define HDIM    128
#define SEQ     4800

#if defined(__HIP_DEVICE_COMPILE__) && __has_builtin(__builtin_amdgcn_tensor_load_to_lds)
#define USE_TDM 1
#endif

#ifdef USE_TDM
// ---------------------------------------------------------------------------
// Tensor Data Mover: async 2-D bf16 tile load (global -> LDS).
// D# built per CDNA5 ISA 8.3/8.4: group0 {count=1, lds_addr, global_addr, type=2},
// group1 {data_size=2B, tensor_dim0=tile_d0, tensor_dim1, tile_dim0/1,
//         tensor_dim0_stride}.  tensor_dim1 < tile_d1 gives zero-fill rows.
// ---------------------------------------------------------------------------
__device__ __forceinline__ void tdm_load_2d_bf16(void* lds_ptr, const bf16* gp,
                                                 uint tile_d0, uint tile_d1,
                                                 uint tensor_d1, uint stride_elems) {
    unsigned long long ga = (unsigned long long)(uintptr_t)gp;
    uint lds_byte = (uint)(uintptr_t)lds_ptr;
    u32x4 g0;
    g0[0] = 1u;                                            // count=1, user mode
    g0[1] = lds_byte;                                      // lds_addr
    g0[2] = (uint)(ga & 0xffffffffu);                      // global_addr[31:0]
    g0[3] = (uint)((ga >> 32) & 0x1ffffffu) | (2u << 30);  // global_addr[56:32] | type=2
    i32x8 g1;
    g1[0] = (int)(1u << 16);                               // data_size = 2 bytes
    g1[1] = (int)((tile_d0 & 0xffffu) << 16);              // tensor_dim0[15:0]
    g1[2] = (int)(((tile_d0 >> 16) & 0xffffu) | ((tensor_d1 & 0xffffu) << 16));
    g1[3] = (int)(((tensor_d1 >> 16) & 0xffffu) | ((tile_d0 & 0xffffu) << 16)); // tile_dim0
    g1[4] = (int)(tile_d1 & 0xffffu);                      // tile_dim1 (tile_dim2 = 0)
    g1[5] = (int)stride_elems;                             // tensor_dim0_stride[31:0]
    g1[6] = 0;
    g1[7] = 0;
    i32x4 z4 = {0, 0, 0, 0};
#if __clang_major__ >= 23
    i32x8 z8 = {0, 0, 0, 0, 0, 0, 0, 0};
    __builtin_amdgcn_tensor_load_to_lds(g0, g1, z4, z4, z8, 0);
#else
    __builtin_amdgcn_tensor_load_to_lds(g0, g1, z4, z4, 0);
#endif
}
#endif

// ---------------------------------------------------------------------------
// fp32 -> bf16 convert (also used for the [out][in] weights: that layout is
// already the B-fragment layout [n][k] for A*W^T, so no transpose needed)
// ---------------------------------------------------------------------------
__global__ __launch_bounds__(256) void cvt_bf16_kernel(const float* __restrict__ src,
                                                       bf16* __restrict__ dst, int n) {
    int i = blockIdx.x * 256 + threadIdx.x;
    if (i < n) dst[i] = (bf16)src[i];
}

// ---------------------------------------------------------------------------
// WMMA GEMM: C[M,N] = A[M,K] * B[N,K]^T + bias[N]
// 128 threads (4 waves).  Block tile 64(M)x128(N), wave tile 16x128, K-step 32.
// A/B tiles staged by TDM (double-buffered) when available.
// ---------------------------------------------------------------------------
__global__ __launch_bounds__(128) void gemm_bf16_kernel(
    const bf16* __restrict__ A, const bf16* __restrict__ B,
    const float* __restrict__ bias, bf16* __restrict__ outb,
    float* __restrict__ outf, int M, int N, int K, int out_f32) {

    __shared__ __align__(16) bf16 As[2][64 * 32];    // [m][k]
    __shared__ __align__(16) bf16 Bs[2][128 * 32];   // [n][k]

    const int tid  = threadIdx.x;
    const int lane = tid & 31;
    const int wv   = tid >> 5;
    const int hi   = (lane >> 4) & 1;
    const int ln   = lane & 15;
    const int m0   = blockIdx.y * 64;
    const int n0   = blockIdx.x * 128;

    v8f acc[8];
#pragma unroll
    for (int t = 0; t < 8; ++t)
#pragma unroll
        for (int r = 0; r < 8; ++r) acc[t][r] = 0.f;

    const int nst = K / 32;
#ifdef USE_TDM
    if (wv == 0) {
        tdm_load_2d_bf16(&As[0][0], A + (size_t)m0 * K, 32, 64, 64, (uint)K);
        tdm_load_2d_bf16(&Bs[0][0], B + (size_t)n0 * K, 32, 128, 128, (uint)K);
    }
#endif
    for (int i = 0; i < nst; ++i) {
        const int bsel = i & 1;
#ifdef USE_TDM
        if (wv == 0) {
            if (i + 1 < nst) {
                int kn = (i + 1) * 32;
                tdm_load_2d_bf16(&As[bsel ^ 1][0], A + (size_t)m0 * K + kn, 32, 64, 64, (uint)K);
                tdm_load_2d_bf16(&Bs[bsel ^ 1][0], B + (size_t)n0 * K + kn, 32, 128, 128, (uint)K);
                __builtin_amdgcn_s_wait_tensorcnt(2);   // current pair done; next pair in flight
            } else {
                __builtin_amdgcn_s_wait_tensorcnt(0);
            }
        }
        __syncthreads();
#else
        const int k0 = i * 32;
#pragma unroll
        for (int i2 = tid; i2 < 256; i2 += 128) {
            int r = i2 >> 2, c4 = i2 & 3;
            reinterpret_cast<uint4*>(&As[bsel][0])[i2] =
                reinterpret_cast<const uint4*>(A + (size_t)(m0 + r) * K + k0)[c4];
        }
#pragma unroll
        for (int i2 = tid; i2 < 512; i2 += 128) {
            int n = i2 >> 2, c4 = i2 & 3;
            reinterpret_cast<uint4*>(&Bs[bsel][0])[i2] =
                reinterpret_cast<const uint4*>(B + (size_t)(n0 + n) * K + k0)[c4];
        }
        __syncthreads();
#endif
        // A fragment (16x32): row = ln, K runs {hi*16+j}
        v16bf a;
        const bf16* ap = &As[bsel][(wv * 16 + ln) * 32];
#pragma unroll
        for (int j = 0; j < 8; ++j) {
            a[j]     = ap[hi * 16 + j];
            a[j + 8] = ap[hi * 16 + 8 + j];
        }
#pragma unroll
        for (int t = 0; t < 8; ++t) {
            v16bf bf;
            const bf16* bp = &Bs[bsel][(t * 16 + ln) * 32 + hi * 16];
#pragma unroll
            for (int j = 0; j < 16; ++j) bf[j] = bp[j];
            acc[t] = __builtin_amdgcn_wmma_f32_16x16x32_bf16(
                false, a, false, bf, (short)0, acc[t], false, false);
        }
        __syncthreads();
    }

#pragma unroll
    for (int t = 0; t < 8; ++t) {
        int col = n0 + t * 16 + ln;
        float bv = bias[col];
#pragma unroll
        for (int r = 0; r < 8; ++r) {
            int row = m0 + wv * 16 + hi * 8 + r;
            float v = acc[t][r] + bv;
            if (out_f32) outf[(size_t)row * N + col] = v;
            else         outb[(size_t)row * N + col] = (bf16)v;
        }
    }
}

// ---------------------------------------------------------------------------
// RMSNorm + 3D axial RoPE (in place on bf16 rows). One block per token.
// ---------------------------------------------------------------------------
__global__ __launch_bounds__(256) void norm_rope_kernel(
    bf16* __restrict__ Qb, bf16* __restrict__ Kb,
    const float* __restrict__ nq, const float* __restrict__ nk,
    const float* __restrict__ fc, const float* __restrict__ fs,
    const int* __restrict__ grid_sizes) {

    __shared__ float red[256];
    const int s   = blockIdx.x;
    const int tid = threadIdx.x;

    const int Hh = grid_sizes[1], Ww = grid_sizes[2];
    const int hw = Hh * Ww;
    const int f  = s / hw;
    const int rm = s - f * hw;
    const int hh = rm / Ww;
    const int wp = rm - hh * Ww;

    for (int pass = 0; pass < 2; ++pass) {
        bf16* X = pass ? Kb : Qb;
        const float* nw = pass ? nk : nq;

        float acc = 0.f;
        for (int i = tid; i < DIM; i += 256) {
            float v = (float)X[(size_t)s * DIM + i];
            acc += v * v;
        }
        red[tid] = acc;
        __syncthreads();
        for (int st = 128; st > 0; st >>= 1) {
            if (tid < st) red[tid] += red[tid + st];
            __syncthreads();
        }
        float rms = rsqrtf(red[0] / (float)DIM + 1e-6f);
        __syncthreads();

        for (int p2 = tid; p2 < NHEADS * 64; p2 += 256) {
            int head = p2 >> 6;
            int p    = p2 & 63;
            size_t idx = (size_t)s * DIM + head * HDIM + 2 * p;
            float re = (float)X[idx]     * rms * nw[head * HDIM + 2 * p];
            float im = (float)X[idx + 1] * rms * nw[head * HDIM + 2 * p + 1];
            int pos = (p < 22) ? f : ((p < 43) ? hh : wp);
            float cs = fc[pos * 64 + p];
            float sn = fs[pos * 64 + p];
            X[idx]     = (bf16)(re * cs - im * sn);
            X[idx + 1] = (bf16)(re * sn + im * cs);
        }
        __syncthreads();
    }
}

// ---------------------------------------------------------------------------
// Flash attention. grid = (SEQ/64, NHEADS), 128 threads (4 waves).
// K tile staged by TDM (zero-fill handles the seq tail); V staged manually
// (needs transpose).  Online softmax in half-wave shuffles.
// ---------------------------------------------------------------------------
__global__ __launch_bounds__(128) void attn_kernel(
    const bf16* __restrict__ Q, const bf16* __restrict__ K,
    const bf16* __restrict__ V, const int* __restrict__ seq_lens,
    bf16* __restrict__ O) {

    __shared__ __align__(16) bf16 Kt[32 * 128];      // [key][dim]
    __shared__ __align__(16) bf16 Vt[128 * 32];      // [dim][key] (transposed)
    __shared__ __align__(16) bf16 Pls[4 * 16 * 32];  // per-wave P tile

    const int tid  = threadIdx.x;
    const int lane = tid & 31;
    const int wv   = tid >> 5;
    const int hi   = (lane >> 4) & 1;
    const int ln   = lane & 15;
    const int head = blockIdx.y;
    const int s0   = blockIdx.x * 64;
    const int len  = seq_lens[0];
    const float rscale = 0.08838834764831845f;       // 1/sqrt(128)

    v16bf qf[4];
#pragma unroll
    for (int kc = 0; kc < 4; ++kc) {
        const bf16* qp = Q + (size_t)(s0 + wv * 16 + ln) * DIM + head * HDIM + kc * 32;
#pragma unroll
        for (int j = 0; j < 8; ++j) {
            qf[kc][j]     = qp[hi * 16 + j];
            qf[kc][j + 8] = qp[hi * 16 + 8 + j];
        }
    }

    v8f o[8];
    float m[8], l[8];
#pragma unroll
    for (int t = 0; t < 8; ++t)
#pragma unroll
        for (int r = 0; r < 8; ++r) o[t][r] = 0.f;
#pragma unroll
    for (int r = 0; r < 8; ++r) { m[r] = -1e30f; l[r] = 0.f; }

    const int nkb = (len + 31) >> 5;
    for (int kb = 0; kb < nkb; ++kb) {
#ifdef USE_TDM
        if (wv == 0)
            tdm_load_2d_bf16(&Kt[0], K + (size_t)(kb * 32) * DIM + head * HDIM,
                             128, 32, (uint)(len - kb * 32), (uint)DIM);
        for (int i = tid; i < 4096; i += 128) {
            int key = i >> 7, d = i & 127;
            int gk = kb * 32 + key;
            Vt[d * 32 + key] = (gk < len) ? V[(size_t)gk * DIM + head * HDIM + d] : (bf16)0.f;
        }
        if (wv == 0) __builtin_amdgcn_s_wait_tensorcnt(0);
        __syncthreads();
#else
        for (int i = tid; i < 4096; i += 128) {
            int key = i >> 7, d = i & 127;
            int gk = kb * 32 + key;
            bf16 kv = (bf16)0.f, vv = (bf16)0.f;
            if (gk < len) {
                kv = K[(size_t)gk * DIM + head * HDIM + d];
                vv = V[(size_t)gk * DIM + head * HDIM + d];
            }
            Kt[key * 128 + d] = kv;
            Vt[d * 32 + key]  = vv;
        }
        __syncthreads();
#endif
        // S = Q * K^T (two 16x16 tiles over 32 keys)
        v8f st[2];
#pragma unroll
        for (int c = 0; c < 2; ++c) {
            v8f s = {};
            int kk = c * 16 + ln;
#pragma unroll
            for (int kc = 0; kc < 4; ++kc) {
                v16bf b;
                const bf16* kp = &Kt[kk * 128 + kc * 32 + hi * 16];
#pragma unroll
                for (int j = 0; j < 16; ++j) b[j] = kp[j];
                s = __builtin_amdgcn_wmma_f32_16x16x32_bf16(
                    false, qf[kc], false, b, (short)0, s, false, false);
            }
            bool valid = (kb * 32 + kk) < len;
#pragma unroll
            for (int r = 0; r < 8; ++r)
                st[c][r] = valid ? s[r] * rscale : -1e30f;
        }

        // online softmax per row (row lives in one 16-lane half)
#pragma unroll
        for (int r = 0; r < 8; ++r) {
            float mx = fmaxf(st[0][r], st[1][r]);
            for (int off = 8; off >= 1; off >>= 1)
                mx = fmaxf(mx, __shfl_xor(mx, off, 32));
            float nm = fmaxf(m[r], mx);
            float al = __expf(m[r] - nm);
            float p0 = __expf(st[0][r] - nm);
            float p1 = __expf(st[1][r] - nm);
            float rs = p0 + p1;
            for (int off = 8; off >= 1; off >>= 1)
                rs += __shfl_xor(rs, off, 32);
            l[r] = l[r] * al + rs;
            m[r] = nm;
#pragma unroll
            for (int t = 0; t < 8; ++t) o[t][r] *= al;
            int prow = hi * 8 + r;
            Pls[wv * 512 + prow * 32 + ln]      = (bf16)p0;
            Pls[wv * 512 + prow * 32 + 16 + ln] = (bf16)p1;
        }
        asm volatile("s_wait_dscnt 0" ::: "memory");

        // O += P * V
        v16bf pf;
        const bf16* pp = &Pls[wv * 512 + ln * 32];
#pragma unroll
        for (int j = 0; j < 8; ++j) {
            pf[j]     = pp[hi * 16 + j];
            pf[j + 8] = pp[hi * 16 + 8 + j];
        }
#pragma unroll
        for (int t = 0; t < 8; ++t) {
            v16bf b;
            const bf16* vp = &Vt[(t * 16 + ln) * 32 + hi * 16];
#pragma unroll
            for (int j = 0; j < 16; ++j) b[j] = vp[j];
            o[t] = __builtin_amdgcn_wmma_f32_16x16x32_bf16(
                false, pf, false, b, (short)0, o[t], false, false);
        }
        __syncthreads();
    }

#pragma unroll
    for (int r = 0; r < 8; ++r) {
        float inv = 1.f / l[r];
        int row = s0 + wv * 16 + hi * 8 + r;
#pragma unroll
        for (int t = 0; t < 8; ++t)
            O[(size_t)row * DIM + head * HDIM + t * 16 + ln] = (bf16)(o[t][r] * inv);
    }
}

// ---------------------------------------------------------------------------
// launch
// ---------------------------------------------------------------------------
extern "C" void kernel_launch(void* const* d_in, const int* in_sizes, int n_in,
                              void* d_out, int out_size, void* d_ws, size_t ws_size,
                              hipStream_t stream) {
    const float* x    = (const float*)d_in[0];
    const int*   slen = (const int*)d_in[1];
    const int*   gsz  = (const int*)d_in[2];
    const float* fc   = (const float*)d_in[3];
    const float* fs   = (const float*)d_in[4];
    const float* q_w  = (const float*)d_in[5];
    const float* q_b  = (const float*)d_in[6];
    const float* k_w  = (const float*)d_in[7];
    const float* k_b  = (const float*)d_in[8];
    const float* v_w  = (const float*)d_in[9];
    const float* v_b  = (const float*)d_in[10];
    const float* o_w  = (const float*)d_in[11];
    const float* o_b  = (const float*)d_in[12];
    const float* nq   = (const float*)d_in[13];
    const float* nk   = (const float*)d_in[14];

    const size_t SZ_ROW = (size_t)SEQ * DIM * sizeof(bf16);
    const size_t SZ_W   = (size_t)DIM * DIM * sizeof(bf16);

    char* p = (char*)d_ws;
    bf16* xb = (bf16*)p;  p += SZ_ROW;
    bf16* wq = (bf16*)p;  p += SZ_W;
    bf16* wk = (bf16*)p;  p += SZ_W;
    bf16* wv = (bf16*)p;  p += SZ_W;
    bf16* wo = (bf16*)p;  p += SZ_W;
    bf16* Qb = (bf16*)p;  p += SZ_ROW;
    bf16* Kb = (bf16*)p;  p += SZ_ROW;
    bf16* Vb = (bf16*)p;  p += SZ_ROW;
    bf16* Ab = (bf16*)p;  p += SZ_ROW;

    const int nElems = SEQ * DIM;
    const int nW     = DIM * DIM;
    cvt_bf16_kernel<<<(nElems + 255) / 256, 256, 0, stream>>>(x, xb, nElems);
    cvt_bf16_kernel<<<(nW + 255) / 256, 256, 0, stream>>>(q_w, wq, nW);
    cvt_bf16_kernel<<<(nW + 255) / 256, 256, 0, stream>>>(k_w, wk, nW);
    cvt_bf16_kernel<<<(nW + 255) / 256, 256, 0, stream>>>(v_w, wv, nW);
    cvt_bf16_kernel<<<(nW + 255) / 256, 256, 0, stream>>>(o_w, wo, nW);

    dim3 ggrid(DIM / 128, SEQ / 64);  // (12, 75)
    gemm_bf16_kernel<<<ggrid, 128, 0, stream>>>(xb, wq, q_b, Qb, nullptr, SEQ, DIM, DIM, 0);
    gemm_bf16_kernel<<<ggrid, 128, 0, stream>>>(xb, wk, k_b, Kb, nullptr, SEQ, DIM, DIM, 0);
    gemm_bf16_kernel<<<ggrid, 128, 0, stream>>>(xb, wv, v_b, Vb, nullptr, SEQ, DIM, DIM, 0);

    norm_rope_kernel<<<SEQ, 256, 0, stream>>>(Qb, Kb, nq, nk, fc, fs, gsz);

    dim3 agrid(SEQ / 64, NHEADS);     // (75, 12)
    attn_kernel<<<agrid, 128, 0, stream>>>(Qb, Kb, Vb, slen, Ab);

    gemm_bf16_kernel<<<ggrid, 128, 0, stream>>>(Ab, wo, o_b, nullptr, (float*)d_out, SEQ, DIM, DIM, 1);
}
